// RNN_29703993819849
// MI455X (gfx1250) — compile-verified, matrix operand order checked by reading, not run on hardware
//
#include <hip/hip_runtime.h>
#include <hip/hip_bf16.h>

// MI455X (gfx1250) LSTM: latency-bound sequential scan.
// Single wave32, W_hh register-resident as WMMA B tiles, recurrence via
// v_wmma_f32_16x16x32_f16 in 12 independent accumulator chains (extraction
// scheduled far from producers to kill WMMA->VALU hazard NOPs), native
// v_tanh_f32 gates, software-pipelined A operand through LDS (in-order,
// barrier-free), scalarized+pipelined x loads, global_prefetch streaming.

typedef __attribute__((ext_vector_type(16))) _Float16 v16h;
typedef __attribute__((ext_vector_type(8)))  _Float16 v8h;
typedef __attribute__((ext_vector_type(8)))  float    v8f;

#define SEQ 131072
#define HID 48
#define NT  12   // 192 gate rows / 16

// CDNA5 native tanh (TRANS op). ISA: 1 independent op needed before the
// result may be consumed -> s_delay_alu TRANS32_DEP_1 inside the asm.
__device__ __forceinline__ float tanh_hw(float v) {
  float r;
  asm volatile("v_tanh_f32 %0, %1\n\t"
               "s_delay_alu instid0(TRANS32_DEP_1)"
               : "=v"(r) : "v"(v));
  return r;
}
__device__ __forceinline__ float sig_hw(float v) {
  // sigmoid(x) = 0.5 + 0.5*tanh(x/2)
  return fmaf(0.5f, tanh_hw(0.5f * v), 0.5f);
}

__global__ __launch_bounds__(32) void lstm_gfx1250(
    const float* __restrict__ x,    const float* __restrict__ Wih,
    const float* __restrict__ Whh,  const float* __restrict__ bih,
    const float* __restrict__ bhh,  const float* __restrict__ Wout,
    const float* __restrict__ bout, const int*   __restrict__ plen,
    float* __restrict__ out)
{
  // hbuf[0..47]   : h (f16) in A-operand memory order
  // hbuf[48..63]  : permanent zeros (K padding 48->64)
  // hbuf[128..175]: dump zone for the duplicate lane group's stores
  __shared__ __align__(16) _Float16 hbuf[256];

  const int lane = threadIdx.x;   // 0..31
  const int nl   = lane & 15;     // N index within tile / hidden sub-index
  const int grp  = lane >> 4;     // half-wave group (K split per WMMA layout)

  for (int i = lane; i < 256; i += 32) hbuf[i] = (_Float16)0.0f;

  // ---- B = W_hh^T, preloaded once into registers, f16 ----
  // 16-bit operand layout (ISA 7.12.2): element e of v16h holds
  // k_local = (e<8 ? e : e+8) + 8*grp ; B[k][n] = Whh[n][k].
  v16h B[NT][2];
#pragma unroll
  for (int p = 0; p < NT; ++p) {
    const int n = p * 16 + nl;
#pragma unroll
    for (int q = 0; q < 2; ++q) {
      v16h b;
#pragma unroll
      for (int e = 0; e < 16; ++e) {
        const int k = 32 * q + (e < 8 ? e : e + 8) + 8 * grp;
        b[e] = (_Float16)((k < HID) ? Whh[n * HID + k] : 0.0f);
      }
      B[p][q] = b;
    }
  }

  // per-lane gate constants: bias(b_ih+b_hh) and the K=2 input projection
  float wx0[NT], wx1[NT], bb[NT];
#pragma unroll
  for (int p = 0; p < NT; ++p) {
    const int j = p * 16 + nl;
    wx0[p] = Wih[2 * j];
    wx1[p] = Wih[2 * j + 1];
    bb[p]  = bih[j] + bhh[j];
  }
  // output head: lanes of grp 1 hold duplicated h -> weight 0 there
  const float wo0 = (grp == 0) ? Wout[nl]      : 0.0f;
  const float wo1 = (grp == 0) ? Wout[16 + nl] : 0.0f;
  const float wo2 = (grp == 0) ? Wout[32 + nl] : 0.0f;
  const float bo  = bout[0];

  float cc[3] = {0.f, 0.f, 0.f};   // cell state, f32; lane nl holds idx u*16+nl
  float hh[3] = {0.f, 0.f, 0.f};
  __syncthreads();                 // fence the one-time hbuf zero-init only

  const v8h* hb     = (const v8h*)hbuf;       // 16B-aligned LDS views
  const int  stbase = nl + (grp ? 128 : 0);   // grp1 stores diverted to dump zone

  // A operand, software-pipelined across steps. Initial load of the zeroed
  // buffer == the h0 = 0 initial state.
  v16h A1, A2;
  {
    v8h lo0 = hb[0 + grp], hi0 = hb[2 + grp];
    v8h lo1 = hb[4 + grp], hi1 = hb[6 + grp];
    A1 = __builtin_shufflevector(lo0, hi0, 0,1,2,3,4,5,6,7,8,9,10,11,12,13,14,15);
    A2 = __builtin_shufflevector(lo1, hi1, 0,1,2,3,4,5,6,7,8,9,10,11,12,13,14,15);
  }

  auto cell = [&](float xa, float xb) {
    // Phase 0: independent VALU filler (covers A-load latency from prev step):
    // per-gate base = bias + W_ih @ x_t.
    float base[NT];
#pragma unroll
    for (int p = 0; p < NT; ++p)
      base[p] = bb[p] + wx0[p] * xa + wx1[p] * xb;

    // Phase 1: 12 independent accumulator chains, 2 WMMAs each.
    const v8f z = {0.f, 0.f, 0.f, 0.f, 0.f, 0.f, 0.f, 0.f};
    v8f acc[NT];
#pragma unroll
    for (int p = 0; p < NT; ++p)
      acc[p] = __builtin_amdgcn_wmma_f32_16x16x32_f16(false, A1, false, B[p][0],
                                                      (short)0, z, false, false);
#pragma unroll
    for (int p = 0; p < NT; ++p)
      acc[p] = __builtin_amdgcn_wmma_f32_16x16x32_f16(false, A2, false, B[p][1],
                                                      (short)0, acc[p], false, false);

    // Phase 2: extraction in producer order -> each read sits far from its
    // WMMA, so the 4-slot WMMA->VALU hazard window is filled by real work.
    float g[NT];
#pragma unroll
    for (int p = 0; p < NT; ++p)
      g[p] = acc[p][0] + base[p];

    // Phase 3: gates [i | f | g | o] = tiles [0..2 | 3..5 | 6..8 | 9..11].
#pragma unroll
    for (int u = 0; u < 3; ++u) {
      const float iv = sig_hw(g[u]);
      const float fv = sig_hw(g[u + 3]);
      const float gv = tanh_hw(g[u + 6]);
      const float ov = sig_hw(g[u + 9]);
      cc[u] = fv * cc[u] + iv * gv;
      hh[u] = ov * tanh_hw(cc[u]);
      hbuf[stbase + u * 16] = (_Float16)hh[u];
    }

    // Phase 4: issue next step's A loads now (after the h stores; single-wave
    // LDS is in-order so the RAW through hbuf is safe without a barrier).
    v8h lo0 = hb[0 + grp], hi0 = hb[2 + grp];
    v8h lo1 = hb[4 + grp], hi1 = hb[6 + grp];
    A1 = __builtin_shufflevector(lo0, hi0, 0,1,2,3,4,5,6,7,8,9,10,11,12,13,14,15);
    A2 = __builtin_shufflevector(lo1, hi1, 0,1,2,3,4,5,6,7,8,9,10,11,12,13,14,15);
  };

  auto project = [&]() -> float {
    float p = hh[0] * wo0 + hh[1] * wo1 + hh[2] * wo2;
#pragma unroll
    for (int off = 16; off > 0; off >>= 1) p += __shfl_xor(p, off, 32);
    return p + bo;
  };

  // ---------------- encoder: sequential scan over SEQ steps ----------------
  float nx0 = x[0], nx1 = x[1];
  for (int t = 0; t < SEQ; ++t) {
    const float xa = nx0, xb = nx1;
    const int tn = (t + 1 < SEQ) ? t + 1 : t;       // pipeline next x by 1 step
    nx0 = x[2 * tn];
    nx1 = x[2 * tn + 1];
    const int pf = (t + 40 < SEQ) ? t + 40 : t;     // stream-prefetch ahead
    __builtin_prefetch(&x[2 * pf], 0, 3);
    cell(xa, xb);
  }

  // ---------------- decoder: 64 autoregressive steps ----------------
  float y = project();                  // out0 from final encoder h
  if (lane == 0) out[0] = y;
  const int P = plen[0];
  float v0 = x[2 * SEQ - 1], v1 = y;    // last raw scalar, then predictions
  for (int s = 1; s < P; ++s) {
    cell(v0, v1);
    y = project();
    if (lane == 0) out[s] = y;
    v0 = v1;
    v1 = y;
  }
}

extern "C" void kernel_launch(void* const* d_in, const int* in_sizes, int n_in,
                              void* d_out, int out_size, void* d_ws, size_t ws_size,
                              hipStream_t stream) {
  const float* x    = (const float*)d_in[0];
  const float* Wih  = (const float*)d_in[1];
  const float* Whh  = (const float*)d_in[2];
  const float* bih  = (const float*)d_in[3];
  const float* bhh  = (const float*)d_in[4];
  const float* Wout = (const float*)d_in[5];
  const float* bout = (const float*)d_in[6];
  const int*   plen = (const int*)d_in[7];
  float* out = (float*)d_out;
  (void)in_sizes; (void)n_in; (void)d_ws; (void)ws_size; (void)out_size;

  lstm_gfx1250<<<1, 32, 0, stream>>>(x, Wih, Whh, bih, bhh, Wout, bout, plen, out);
}
